// MLSWATransformerBlock_57148834840627
// MI455X (gfx1250) — compile-verified, matrix operand order checked by reading, not run on hardware
//
#include <hip/hip_runtime.h>
#include <math.h>

typedef __bf16 bf16_t;
typedef __attribute__((ext_vector_type(16))) __bf16 v16bf;
typedef __attribute__((ext_vector_type(8)))  __bf16 v8bf;
typedef __attribute__((ext_vector_type(8)))  float  v8f;

#define B_    2
#define N_    2048
#define D_    1024
#define H_    16
#define HD_   64
#define DFF_  4096
#define M_ROWS (B_ * N_)   /* 4096 token rows */

// ---------------------------------------------------------------------------
// CDNA5 WMMA helpers (wave32). 16x16x32 bf16, f32 accumulate.
// A-frag (16x32): lanes 0-15 & 16-31 both hold M=lane&15; per-lane 16 bf16,
// element e -> VGPR e/2, K = ((v&3)*2) + ((v>>2)*16) + (lane>>4)*8 + (e&1).
// D-frag (16x16 f32): lane half encodes M: m = (lane>>4)*8 + r, n = lane&15.
// ---------------------------------------------------------------------------
__device__ __forceinline__ int frag_k(int e, int half) {
  int v = e >> 1, s = e & 1;
  return ((v & 3) * 2) + ((v >> 2) * 16) + half * 8 + s;
}

__device__ __forceinline__ v8f wmma_bf16(v16bf a, v16bf b, v8f c) {
  return __builtin_amdgcn_wmma_f32_16x16x32_bf16(
      /*neg_a=*/false, a, /*neg_b=*/false, b,
      /*c_mod=*/(short)0, c, /*reuse_a=*/false, /*reuse_b=*/false);
}

__device__ __forceinline__ v8f zero_v8f() {
  v8f z;
#pragma unroll
  for (int e = 0; e < 8; ++e) z[e] = 0.0f;
  return z;
}

// ---------------------------------------------------------------------------
// CDNA5 async global->LDS copy (ASYNCcnt-tracked); inline asm is the
// toolchain-portable path. Generic LDS pointer's low 32 bits = LDS address.
// ---------------------------------------------------------------------------
__device__ __forceinline__ void async_load_b128(void* lds, const void* gptr) {
  unsigned lds_addr = (unsigned)(unsigned long long)lds;
  unsigned long long gaddr = (unsigned long long)gptr;
  asm volatile("global_load_async_to_lds_b128 %0, %1, off"
               :: "v"(lds_addr), "v"(gaddr) : "memory");
}

__device__ __forceinline__ void wait_asynccnt0() {
  asm volatile("s_wait_asynccnt 0x0" ::: "memory");
}

// ---------------------------------------------------------------------------
// LayerNorm (f32 in -> bf16 out), one block per token row, D = 1024.
// ---------------------------------------------------------------------------
__global__ __launch_bounds__(256) void ln_bf16_kernel(
    const float* __restrict__ x, const float* __restrict__ g,
    const float* __restrict__ bta, bf16_t* __restrict__ out) {
  const int row = blockIdx.x;
  const int tid = threadIdx.x;
  const float* xr = x + (size_t)row * D_;
  float s = 0.0f, s2 = 0.0f;
#pragma unroll
  for (int i = 0; i < D_ / 256; ++i) {
    float v = xr[tid + i * 256];
    s += v; s2 += v * v;
  }
#pragma unroll
  for (int off = 1; off < 32; off <<= 1) {
    s  += __shfl_xor(s,  off, 32);
    s2 += __shfl_xor(s2, off, 32);
  }
  __shared__ float wsum[8], wsq[8];
  const int wave = tid >> 5, lane = tid & 31;
  if (lane == 0) { wsum[wave] = s; wsq[wave] = s2; }
  __syncthreads();
  if (wave == 0) {
    float a = (lane < 8) ? wsum[lane] : 0.0f;
    float b = (lane < 8) ? wsq[lane]  : 0.0f;
#pragma unroll
    for (int off = 1; off < 8; off <<= 1) {
      a += __shfl_xor(a, off, 32);
      b += __shfl_xor(b, off, 32);
    }
    if (lane == 0) { wsum[0] = a; wsq[0] = b; }
  }
  __syncthreads();
  const float mean = wsum[0] * (1.0f / D_);
  const float var  = wsq[0] * (1.0f / D_) - mean * mean;
  const float inv  = rsqrtf(var + 1e-5f);
#pragma unroll
  for (int i = 0; i < D_ / 256; ++i) {
    const int c = tid + i * 256;
    float v = (xr[c] - mean) * inv * g[c] + bta[c];
    out[(size_t)row * D_ + c] = (bf16_t)v;
  }
}

// ---------------------------------------------------------------------------
// Weight prep: f32 [K][Nn] -> bf16 transposed [Nn][K] (coalesced writes).
// ---------------------------------------------------------------------------
__global__ void cvtT_f32_bf16(const float* __restrict__ in,
                              bf16_t* __restrict__ out, int K, int Nn) {
  int i = blockIdx.x * blockDim.x + threadIdx.x;
  const int total = K * Nn;
  const int stride = gridDim.x * blockDim.x;
  for (; i < total; i += stride) {
    const int n = i / K, k = i - n * K;
    out[i] = (bf16_t)in[(size_t)k * Nn + n];
  }
}

// ---------------------------------------------------------------------------
// V pre-transpose: qkv (B,N,3,H,HD) bf16 -> vt (B,H,HD,N) bf16, so the
// attention V tile is a row-major copy (async-LDS stageable, no ds scatter).
// ---------------------------------------------------------------------------
__global__ void transpose_v_kernel(const bf16_t* __restrict__ qkv,
                                   bf16_t* __restrict__ vt) {
  int i = blockIdx.x * blockDim.x + threadIdx.x;
  const int total = B_ * H_ * HD_ * N_;   // 2^22
  const int stride = gridDim.x * blockDim.x;
  for (; i < total; i += stride) {
    const int n  = i & (N_ - 1);
    const int hd = (i >> 11) & (HD_ - 1);
    const int h  = (i >> 17) & (H_ - 1);
    const int b  = i >> 21;
    vt[i] = qkv[(((size_t)(b * N_ + n) * 3 + 2) * H_ + h) * HD_ + hd];
  }
}

// ---------------------------------------------------------------------------
// Tiled bf16 WMMA GEMM: C(MxN) = A(MxK) @ B^T(NxK)^T + bias, fused epilogues.
// Block tile 128x128, BK=64, 8 waves (2x4), wave tile 64x32 (4x2 frags).
// Double-buffered async global->LDS staging overlapped with 16 WMMAs/tile.
// ---------------------------------------------------------------------------
enum { EPI_BF16 = 0, EPI_GELU_BF16 = 1, EPI_RES_F32 = 2 };

template <int EPI>
__global__ __launch_bounds__(256) void gemm_bf16_128x128(
    const bf16_t* __restrict__ A, const bf16_t* __restrict__ Bt,
    const float* __restrict__ bias, float* __restrict__ Cf,
    bf16_t* __restrict__ Cb, const float* __restrict__ resid,
    int M, int Nn, int K) {
  __shared__ bf16_t sA[2][128 * 64];  // [m][k]
  __shared__ bf16_t sB[2][128 * 64];  // [n][k]
  const int tid   = threadIdx.x;
  const int wave  = tid >> 5, lane = tid & 31;
  const int half  = lane >> 4, l16 = lane & 15;
  const int waveM = wave >> 2, waveN = wave & 3;
  const int m0 = blockIdx.y * 128, n0 = blockIdx.x * 128;

  v8f acc[4][2];
#pragma unroll
  for (int im = 0; im < 4; ++im)
#pragma unroll
    for (int jn = 0; jn < 2; ++jn) acc[im][jn] = zero_v8f();

  const int kTiles = K >> 6;

  // 128x64 tile = 8192 bf16 = 1024 x b128; 4 chunks/thread per matrix.
  auto stage_tile = [&](int kt, int buf) {
#pragma unroll
    for (int p = 0; p < 4; ++p) {
      const int c   = tid + p * 256;
      const int row = c >> 3;
      const int col = (c & 7) * 8;
      async_load_b128(&sA[buf][row * 64 + col],
                      &A[(size_t)(m0 + row) * K + kt * 64 + col]);
      async_load_b128(&sB[buf][row * 64 + col],
                      &Bt[(size_t)(n0 + row) * K + kt * 64 + col]);
    }
  };

  stage_tile(0, 0);
  for (int kt = 0; kt < kTiles; ++kt) {
    wait_asynccnt0();      // my async writes into buf[kt&1] have landed
    __syncthreads();       // everyone's have
    if (kt + 1 < kTiles) stage_tile(kt + 1, (kt + 1) & 1);

    const bf16_t* pA = sA[kt & 1];
    const bf16_t* pB = sB[kt & 1];
#pragma unroll
    for (int kk = 0; kk < 2; ++kk) {
      v16bf afr[4], bfr[2];
#pragma unroll
      for (int im = 0; im < 4; ++im) {
        const bf16_t* src = &pA[(waveM * 64 + im * 16 + l16) * 64 + kk * 32];
#pragma unroll
        for (int e = 0; e < 16; ++e) afr[im][e] = src[frag_k(e, half)];
      }
#pragma unroll
      for (int jn = 0; jn < 2; ++jn) {
        const bf16_t* src = &pB[(waveN * 32 + jn * 16 + l16) * 64 + kk * 32];
#pragma unroll
        for (int e = 0; e < 16; ++e) bfr[jn][e] = src[frag_k(e, half)];
      }
#pragma unroll
      for (int im = 0; im < 4; ++im)
#pragma unroll
        for (int jn = 0; jn < 2; ++jn)
          acc[im][jn] = wmma_bf16(afr[im], bfr[jn], acc[im][jn]);
    }
    __syncthreads();       // reads done before this buffer is re-staged
  }

  // Epilogue (D-frag layout: m = half*8 + r, n = l16).
#pragma unroll
  for (int im = 0; im < 4; ++im) {
#pragma unroll
    for (int jn = 0; jn < 2; ++jn) {
#pragma unroll
      for (int r = 0; r < 8; ++r) {
        const int m = m0 + waveM * 64 + im * 16 + half * 8 + r;
        const int n = n0 + waveN * 32 + jn * 16 + l16;
        float v = acc[im][jn][r] + bias[n];
        const size_t idx = (size_t)m * Nn + n;
        if (EPI == EPI_BF16) {
          Cb[idx] = (bf16_t)v;
        } else if (EPI == EPI_GELU_BF16) {
          float gel = 0.5f * v * (1.0f + erff(v * 0.7071067811865475f));
          Cb[idx] = (bf16_t)gel;
        } else {  // EPI_RES_F32
          Cf[idx] = v + resid[idx];
        }
      }
    }
  }
}

// ---------------------------------------------------------------------------
// MLSWA flash attention. Block: one (b, h, 128-query tile), 8 waves; each
// wave owns 16 queries. Streams 64-key blocks with double-buffered async
// global->LDS staging of K and pre-transposed V. Per block: 8 WMMAs for
// S = Q K^T, online softmax (half-wave shuffles), P re-laid via per-wave
// LDS, 8 WMMAs for P·V.
// qkv layout: (B, N, 3, H, HD) bf16; vt layout: (B, H, HD, N) bf16.
// ---------------------------------------------------------------------------
__global__ __launch_bounds__(256) void mlswa_attn_kernel(
    const bf16_t* __restrict__ qkv, const bf16_t* __restrict__ vt,
    const int* __restrict__ pos, bf16_t* __restrict__ out) {
  __shared__ bf16_t sK[2][64 * 64];    // [key_local][hd]
  __shared__ bf16_t sV[2][64 * 64];    // [hd][key_local]
  __shared__ bf16_t sP[8 * 16 * 64];   // per-wave P scratch (A-layout source)

  const int tid  = threadIdx.x;
  const int wave = tid >> 5, lane = tid & 31;
  const int half = lane >> 4, l16 = lane & 15;
  const int b = blockIdx.z, h = blockIdx.y;
  const int i0 = blockIdx.x * 128 + wave * 16;

  const size_t tok = 3 * (size_t)D_;   // per-token stride in qkv elements
  const bf16_t* qbase = qkv + ((size_t)b * N_) * tok + (size_t)h * HD_;
  const bf16_t* kbase = qbase + D_;    // s = 1
  const bf16_t* vtb   = vt + ((size_t)(b * H_ + h)) * HD_ * N_;

  // Q fragments: K-chunks of 32 over HD=64.
  v16bf aq[2];
#pragma unroll
  for (int kk = 0; kk < 2; ++kk) {
    const bf16_t* qr = qbase + (size_t)(i0 + l16) * tok + kk * 32;
#pragma unroll
    for (int e = 0; e < 16; ++e) aq[kk][e] = qr[frag_k(e, half)];
  }

  // Multi-level positions for this lane's 8 rows (D-layout convention).
  int pi0[8], pi1[8], pi2[8];
#pragma unroll
  for (int r = 0; r < 8; ++r) {
    const int i = i0 + half * 8 + r;
    pi0[r] = pos[i]; pi1[r] = pos[N_ + i]; pi2[r] = pos[2 * N_ + i];
  }

  v8f O[4];
#pragma unroll
  for (int f = 0; f < 4; ++f) O[f] = zero_v8f();
  float mrun[8], lrun[8];
#pragma unroll
  for (int r = 0; r < 8; ++r) { mrun[r] = -1e30f; lrun[r] = 0.0f; }

  const float scale = 0.125f;  // 1/sqrt(HD)

  // Stage a 64-key block: K tile 64x64 + V tile 64x64, each 512 x b128,
  // 2 chunks per thread per matrix.
  auto stage_kv = [&](int jt, int buf) {
    const int j0 = jt * 64;
#pragma unroll
    for (int p = 0; p < 2; ++p) {
      const int c = tid + p * 256;
      const int row = c >> 3;          // key for K, hd for V
      const int col = (c & 7) * 8;
      async_load_b128(&sK[buf][row * 64 + col],
                      &kbase[(size_t)(j0 + row) * tok + col]);
      async_load_b128(&sV[buf][row * 64 + col],
                      &vtb[(size_t)row * N_ + j0 + col]);
    }
  };

  const int jTiles = N_ / 64;
  stage_kv(0, 0);
  for (int jt = 0; jt < jTiles; ++jt) {
    wait_asynccnt0();
    __syncthreads();
    if (jt + 1 < jTiles) stage_kv(jt + 1, (jt + 1) & 1);
    const bf16_t* pK = sK[jt & 1];
    const bf16_t* pV = sV[jt & 1];
    const int j0 = jt * 64;

    // S = Q K^T over four 16-key subtiles, K-dim 64 = 2 WMMA steps each.
    v8f S[4];
#pragma unroll
    for (int t = 0; t < 4; ++t) {
      S[t] = zero_v8f();
#pragma unroll
      for (int kk = 0; kk < 2; ++kk) {
        v16bf bk;
        const bf16_t* src = &pK[(t * 16 + l16) * 64 + kk * 32];
#pragma unroll
        for (int e = 0; e < 16; ++e) bk[e] = src[frag_k(e, half)];
        S[t] = wmma_bf16(aq[kk], bk, S[t]);
      }
    }

    // Scale + multi-level sliding-window mask.
#pragma unroll
    for (int t = 0; t < 4; ++t) {
      const int j = j0 + t * 16 + l16;
      const int pj0 = pos[j], pj1 = pos[N_ + j], pj2 = pos[2 * N_ + j];
#pragma unroll
      for (int r = 0; r < 8; ++r) {
        const int d0 = pi0[r] - pj0, d1 = pi1[r] - pj1, d2 = pi2[r] - pj2;
        const bool ok = (d0 < 256 && d0 > -256) || (d1 < 64 && d1 > -64) ||
                        (d2 < 16 && d2 > -16);
        const float v = S[t][r] * scale;
        S[t][r] = ok ? v : -1e9f;
      }
    }

    // Online softmax: row reductions over 16-lane halves.
    float alpha[8];
#pragma unroll
    for (int r = 0; r < 8; ++r) {
      float rm = fmaxf(fmaxf(S[0][r], S[1][r]), fmaxf(S[2][r], S[3][r]));
#pragma unroll
      for (int off = 1; off < 16; off <<= 1)
        rm = fmaxf(rm, __shfl_xor(rm, off, 32));
      const float mnew = fmaxf(mrun[r], rm);
      alpha[r] = __expf(mrun[r] - mnew);
      mrun[r] = mnew;
      float rs = 0.0f;
#pragma unroll
      for (int t = 0; t < 4; ++t) {
        const float p = __expf(S[t][r] - mnew);
        S[t][r] = p;
        rs += p;
      }
#pragma unroll
      for (int off = 1; off < 16; off <<= 1) rs += __shfl_xor(rs, off, 32);
      lrun[r] = lrun[r] * alpha[r] + rs;
    }
#pragma unroll
    for (int f = 0; f < 4; ++f)
#pragma unroll
      for (int r = 0; r < 8; ++r) O[f][r] *= alpha[r];

    // P (D-layout f32, 16x64) -> per-wave LDS -> two A-layout bf16 frags.
    bf16_t* pw = &sP[wave * 1024];
#pragma unroll
    for (int t = 0; t < 4; ++t)
#pragma unroll
      for (int r = 0; r < 8; ++r)
        pw[(half * 8 + r) * 64 + t * 16 + l16] = (bf16_t)S[t][r];

    v16bf ap[2];
#pragma unroll
    for (int kc = 0; kc < 2; ++kc) {
      const bf16_t* src = &pw[l16 * 64 + kc * 32];
#pragma unroll
      for (int e = 0; e < 16; ++e) ap[kc][e] = src[frag_k(e, half)];
    }

    // O += P @ V over 4 hd-tiles of 16, key-dim 64 = 2 WMMA steps each.
#pragma unroll
    for (int f = 0; f < 4; ++f) {
#pragma unroll
      for (int kc = 0; kc < 2; ++kc) {
        v16bf bv;
        const bf16_t* src = &pV[(f * 16 + l16) * 64 + kc * 32];
#pragma unroll
        for (int e = 0; e < 16; ++e) bv[e] = src[frag_k(e, half)];
        O[f] = wmma_bf16(ap[kc], bv, O[f]);
      }
    }
    __syncthreads();   // all reads of this buffer done before re-staging
  }

  // Normalize and store (B, N, D) bf16.
#pragma unroll
  for (int r = 0; r < 8; ++r) lrun[r] = 1.0f / lrun[r];
#pragma unroll
  for (int f = 0; f < 4; ++f) {
#pragma unroll
    for (int r = 0; r < 8; ++r) {
      const int i = i0 + half * 8 + r;
      const int col = h * HD_ + f * 16 + l16;
      out[((size_t)b * N_ + i) * D_ + col] = (bf16_t)(O[f][r] * lrun[r]);
    }
  }
}

// ---------------------------------------------------------------------------
// Launch: LN1 -> cvt+transpose weights -> QKV GEMM -> V transpose ->
//         attention -> out-proj(+residual) -> LN2 -> FF1(+GELU) ->
//         FF2(+residual) -> d_out.
// ---------------------------------------------------------------------------
extern "C" void kernel_launch(void* const* d_in, const int* in_sizes, int n_in,
                              void* d_out, int out_size, void* d_ws,
                              size_t ws_size, hipStream_t stream) {
  (void)in_sizes; (void)n_in; (void)out_size; (void)ws_size;
  const float* x     = (const float*)d_in[0];
  const int*   pos   = (const int*)d_in[1];
  const float* w_qkv = (const float*)d_in[2];
  const float* b_qkv = (const float*)d_in[3];
  const float* w_out = (const float*)d_in[4];
  const float* b_out = (const float*)d_in[5];
  const float* w_ff1 = (const float*)d_in[6];
  const float* b_ff1 = (const float*)d_in[7];
  const float* w_ff2 = (const float*)d_in[8];
  const float* b_ff2 = (const float*)d_in[9];
  const float* g1    = (const float*)d_in[10];
  const float* be1   = (const float*)d_in[11];
  const float* g2    = (const float*)d_in[12];
  const float* be2   = (const float*)d_in[13];
  float* outp = (float*)d_out;

  char* ws = (char*)d_ws;
  auto alloc = [&](size_t bytes) {
    char* p = ws;
    ws += (bytes + 255) & ~(size_t)255;
    return p;
  };
  bf16_t* xn      = (bf16_t*)alloc((size_t)M_ROWS * D_ * 2);
  bf16_t* wqkv_t  = (bf16_t*)alloc((size_t)D_ * 3 * D_ * 2);
  bf16_t* wout_t  = (bf16_t*)alloc((size_t)D_ * D_ * 2);
  bf16_t* wff1_t  = (bf16_t*)alloc((size_t)D_ * DFF_ * 2);
  bf16_t* wff2_t  = (bf16_t*)alloc((size_t)DFF_ * D_ * 2);
  bf16_t* qkvb    = (bf16_t*)alloc((size_t)M_ROWS * 3 * D_ * 2);
  bf16_t* vtb     = (bf16_t*)alloc((size_t)B_ * H_ * HD_ * N_ * 2);
  bf16_t* attnb   = (bf16_t*)alloc((size_t)M_ROWS * D_ * 2);
  float*  x1      = (float*) alloc((size_t)M_ROWS * D_ * 4);
  bf16_t* xn2     = (bf16_t*)alloc((size_t)M_ROWS * D_ * 2);
  bf16_t* hb      = (bf16_t*)alloc((size_t)M_ROWS * DFF_ * 2);

  ln_bf16_kernel<<<M_ROWS, 256, 0, stream>>>(x, g1, be1, xn);
  cvtT_f32_bf16<<<1024, 256, 0, stream>>>(w_qkv, wqkv_t, D_, 3 * D_);
  cvtT_f32_bf16<<<512,  256, 0, stream>>>(w_out, wout_t, D_, D_);
  cvtT_f32_bf16<<<1024, 256, 0, stream>>>(w_ff1, wff1_t, D_, DFF_);
  cvtT_f32_bf16<<<1024, 256, 0, stream>>>(w_ff2, wff2_t, DFF_, D_);

  gemm_bf16_128x128<EPI_BF16>
      <<<dim3(3 * D_ / 128, M_ROWS / 128), 256, 0, stream>>>(
          xn, wqkv_t, b_qkv, nullptr, qkvb, nullptr, M_ROWS, 3 * D_, D_);

  transpose_v_kernel<<<2048, 256, 0, stream>>>(qkvb, vtb);

  mlswa_attn_kernel<<<dim3(N_ / 128, H_, B_), 256, 0, stream>>>(qkvb, vtb,
                                                                pos, attnb);

  gemm_bf16_128x128<EPI_RES_F32>
      <<<dim3(D_ / 128, M_ROWS / 128), 256, 0, stream>>>(
          attnb, wout_t, b_out, x1, nullptr, x, M_ROWS, D_, D_);

  ln_bf16_kernel<<<M_ROWS, 256, 0, stream>>>(x1, g2, be2, xn2);

  gemm_bf16_128x128<EPI_GELU_BF16>
      <<<dim3(DFF_ / 128, M_ROWS / 128), 256, 0, stream>>>(
          xn2, wff1_t, b_ff1, nullptr, hb, nullptr, M_ROWS, DFF_, D_);

  gemm_bf16_128x128<EPI_RES_F32>
      <<<dim3(D_ / 128, M_ROWS / 128), 256, 0, stream>>>(
          hb, wff2_t, b_ff2, outp, nullptr, x1, M_ROWS, D_, DFF_);
}
